// MoE_9844065042869
// MI455X (gfx1250) — matrix operand order repaired and checked by reading, not compiled, not run on hardware
//
#include <hip/hip_runtime.h>
#include <hip/hip_bf16.h>

#define B_   4096
#define D_   4096
#define P_   64
#define E_   8
#define C2_  64   // 2*C
#define K_   64
#define LDSB 72   // bf16 LDS row stride (144 B, keeps 16B alignment of k-runs)

typedef __attribute__((ext_vector_type(16))) __bf16 v16bf;
typedef __attribute__((ext_vector_type(8)))  float  v8f;

__device__ __forceinline__ unsigned short f2bf(float f) {
  // round-to-nearest-even f32 -> bf16 (inputs are finite normals)
  unsigned int u = __float_as_uint(f);
  u += 0x7FFFu + ((u >> 16) & 1u);
  return (unsigned short)(u >> 16);
}
__device__ __forceinline__ unsigned pack_bf2(float a, float b) {
  return (unsigned)f2bf(a) | ((unsigned)f2bf(b) << 16);
}

// One block per batch element: stage x[b] as bf16 in LDS (+f32 routing sums in
// regs), route in f32, stage w_exp[e] as bf16, then 64x64x64 expert GEMM via
// v_wmma_f32_16x16x32_bf16 with ds_load_b128 fragment loads.
__global__ __launch_bounds__(256) void moe_fused_kernel(
    const float* __restrict__ x, const float* __restrict__ w_router,
    const float* __restrict__ w_exp, const float* __restrict__ b_exp,
    float* __restrict__ out, float* __restrict__ ws_partial) {
  __shared__ __bf16 xpb[P_ * LDSB];    // xp[p][k] in bf16 (B fragments)
  __shared__ __bf16 wb[C2_ * LDSB];    // w_exp[e][c][k] in bf16 (A fragments)
  __shared__ float  sred[16 * K_];     // routing partial column sums
  __shared__ float  sfin[K_];          // s[k] = sum_p xp[p][k]
  __shared__ float  sel[E_];
  __shared__ float  gate_sh;
  __shared__ int    idx_sh;
  __shared__ float  red[256];

  const int tid = threadIdx.x;
  const int b   = blockIdx.x;
  const float* __restrict__ xb = x + (size_t)b * D_;

  // ---- Stage x[b]: global b128 -> packed bf16 LDS + routing partials ----
  {
    const int k0 = (tid * 4) & 63;     // same k range for all r
    float rs0 = 0.f, rs1 = 0.f, rs2 = 0.f, rs3 = 0.f;
#pragma unroll
    for (int r = 0; r < 4; ++r) {
      int e4  = tid + r * 256;         // float4 index 0..1023
      int off = e4 * 4;
      float4 v = ((const float4*)xb)[e4];
      int p = off >> 6;                // k == k0 for every r
      unsigned* dst = (unsigned*)&xpb[p * LDSB + k0];
      dst[0] = pack_bf2(v.x, v.y);
      dst[1] = pack_bf2(v.z, v.w);
      rs0 += v.x; rs1 += v.y; rs2 += v.z; rs3 += v.w;
    }
    const int g = tid >> 4;            // 16 thread-groups over p
    float* s = &sred[g * K_ + k0];
    s[0] = rs0; s[1] = rs1; s[2] = rs2; s[3] = rs3;
  }
  __syncthreads();

  // ---- Routing: finalize s[k], select = s . w_router, argmax ----
  if (tid < K_) {
    float s = 0.f;
#pragma unroll
    for (int g = 0; g < 16; ++g) s += sred[g * K_ + tid];
    sfin[tid] = s;
  }
  __syncthreads();
  if (tid < E_) {
    float a = 0.f;
#pragma unroll
    for (int k = 0; k < K_; ++k) a += sfin[k] * w_router[tid * K_ + k];
    sel[tid] = a;
  }
  __syncthreads();
  if (tid == 0) {
    int im = 0; float g = sel[0];
#pragma unroll
    for (int e = 1; e < E_; ++e) { if (sel[e] > g) { g = sel[e]; im = e; } }
    gate_sh = g; idx_sh = im;
  }
  __syncthreads();

  const int eIdx = idx_sh;

  // dispatch row + per-block lb partials (deterministic two-stage reduce)
  if (tid < E_) {
    float g = gate_sh; int im = idx_sh;
    out[2 * B_ + (size_t)b * E_ + tid] = (tid == im && g != 0.0f) ? 1.0f : 0.0f;
    ws_partial[(size_t)b * 16 + tid]     = sel[tid];          // density_proxy
    ws_partial[(size_t)b * 16 + 8 + tid] = (tid == im) ? 1.0f : 0.0f; // density
  }

  // ---- Stage w_exp[eIdx]: coalesced b128 -> packed bf16 LDS ----
  {
    const float* __restrict__ wep = w_exp + (size_t)eIdx * (C2_ * K_);
    const int k0 = (tid * 4) & 63;
#pragma unroll
    for (int r = 0; r < 4; ++r) {
      int e4  = tid + r * 256;
      int off = e4 * 4;
      float4 w = ((const float4*)wep)[e4];
      int row = off >> 6;
      unsigned* dst = (unsigned*)&wb[row * LDSB + k0];
      dst[0] = pack_bf2(w.x, w.y);
      dst[1] = pack_bf2(w.z, w.w);
    }
  }
  __syncthreads();

  // ---- Expert GEMM: z[c,p] = w_exp[e] (64x64) * xp^T (64x64), bf16 WMMA ----
  const int lane = tid & 31, wv = tid >> 5;
  const int c_tile = wv >> 1;                 // waves 0-1 -> ct0 ... 6-7 -> ct3
  const int pt0    = (wv & 1) ? 2 : 0;        // two p-tiles per wave
  const int hi     = lane >> 4;               // half-wave
  const int ln     = lane & 15;

  union Frag { v16bf v; uint4 q[2]; };

  v8f acc0 = {};
  v8f acc1 = {};
  const __bf16* arow = &wb[(c_tile * 16 + ln) * LDSB];
  const __bf16* brow0 = &xpb[(pt0 * 16 + ln) * LDSB];
  const __bf16* brow1 = &xpb[((pt0 + 1) * 16 + ln) * LDSB];
#pragma unroll
  for (int ks = 0; ks < 2; ++ks) {
    const int kbase = ks * 32;
    // A fragment (16x32): lane = row c; elems 0-15 -> K=kbase+hi*16+0..15
    // (two contiguous 16B runs)
    Frag A;
    A.q[0] = *(const uint4*)&arow[kbase + hi * 16];
    A.q[1] = *(const uint4*)&arow[kbase + hi * 16 + 8];
    // B fragment (32x16): lane = col p; elems 0-15 -> K=kbase+hi*16+0..15
    const int kb = kbase + hi * 16;
    Frag Bf0, Bf1;
    Bf0.q[0] = *(const uint4*)&brow0[kb];
    Bf0.q[1] = *(const uint4*)&brow0[kb + 8];
    Bf1.q[0] = *(const uint4*)&brow1[kb];
    Bf1.q[1] = *(const uint4*)&brow1[kb + 8];
    acc0 = __builtin_amdgcn_wmma_f32_16x16x32_bf16(false, A.v, false, Bf0.v,
                                                   (short)0, acc0, false, false);
    acc1 = __builtin_amdgcn_wmma_f32_16x16x32_bf16(false, A.v, false, Bf1.v,
                                                   (short)0, acc1, false, false);
  }

  // ---- bias + cube + channel-group / position reduction ----
  float part = 0.f;
  const float* __restrict__ bep = b_exp + eIdx * C2_ + c_tile * 16 + hi * 8;
#pragma unroll
  for (int j = 0; j < 8; ++j) {
    float bb = bep[j];
    float v0 = acc0[j] + bb;
    float v1 = acc1[j] + bb;
    part += v0 * v0 * v0 + v1 * v1 * v1;
  }
  // waves 0-3 (threads 0..127) hold channel group 0; waves 4-7 group 1
  red[tid] = part;
  __syncthreads();
#pragma unroll
  for (int s = 64; s > 0; s >>= 1) {
    if ((tid & 127) < s) red[tid] += red[tid + s];
    __syncthreads();
  }
  if (tid == 0) {
    float g  = gate_sh;
    float o0 = g * red[0];
    float o1 = g * red[128];
    float mx = fmaxf(o0, o1);
    float e0 = __expf(o0 - mx), e1 = __expf(o1 - mx);
    float inv = 1.0f / (e0 + e1);
    out[(size_t)b * 2 + 0] = e0 * inv;
    out[(size_t)b * 2 + 1] = e1 * inv;
  }
}

// Deterministic reduction of per-block lb partials -> scalar lb_loss.
__global__ __launch_bounds__(256) void lb_reduce_kernel(
    const float* __restrict__ ws_partial, float* __restrict__ out) {
  __shared__ float sh[16 * 256];
  __shared__ float fin[16];
  const int tid = threadIdx.x;
  float acc[16];
#pragma unroll
  for (int j = 0; j < 16; ++j) acc[j] = 0.f;
  for (int b = tid; b < B_; b += 256) {
    const float* p = ws_partial + (size_t)b * 16;
#pragma unroll
    for (int j = 0; j < 16; ++j) acc[j] += p[j];
  }
#pragma unroll
  for (int j = 0; j < 16; ++j) sh[j * 256 + tid] = acc[j];
  __syncthreads();
  if (tid < 16) {
    float s = 0.f;
    for (int t = 0; t < 256; ++t) s += sh[tid * 256 + t];
    fin[tid] = s;
  }
  __syncthreads();
  if (tid == 0) {
    const float invB = 1.0f / (float)B_;
    float lb = 0.f;
#pragma unroll
    for (int e = 0; e < E_; ++e)
      lb += (fin[e] * invB) * (fin[8 + e] * invB);
    lb *= (float)(E_ * E_) / (float)E_;   // mean over E, times E^2
    out[2 * B_ + E_ * B_] = lb;
  }
}

extern "C" void kernel_launch(void* const* d_in, const int* in_sizes, int n_in,
                              void* d_out, int out_size, void* d_ws, size_t ws_size,
                              hipStream_t stream) {
  const float* x        = (const float*)d_in[0];
  const float* w_router = (const float*)d_in[1];
  const float* w_exp    = (const float*)d_in[2];
  const float* b_exp    = (const float*)d_in[3];
  float* out = (float*)d_out;
  float* wsf = (float*)d_ws;   // needs B_*16*4 = 256 KB

  moe_fused_kernel<<<B_, 256, 0, stream>>>(x, w_router, w_exp, b_exp, out, wsf);
  lb_reduce_kernel<<<1, 256, 0, stream>>>(wsf, out);
}
